// FBResEEGMoE_77421080478078
// MI455X (gfx1250) — matrix operand order repaired and checked by reading, not compiled, not run on hardware
//
#include <hip/hip_runtime.h>
#include <math.h>

// ---- problem constants ----
#define BSZ 256
#define CC  22
#define TT  1000
#define NB  6
#define EE  6
#define POOL 125
#define NBINS 8       // TT / POOL
#define NCLS 9
#define MNB 48        // M * NB
#define NTAPS 21
#define GTAPS 27      // composed 21 (FIR) * 7 (avg of 3/5/7) filter
#define NTILES 63     // ceil(1000/16)

typedef __attribute__((ext_vector_type(16))) _Float16 v16h;
typedef __attribute__((ext_vector_type(8)))  float    v8f;

// =====================================================================
// Kernel 1: fused FIR + multiscale conv + BN + ReLU + sconv + pool.
// One wave per (time-tile, batch); one V_WMMA_F32_16X16X32_F16 per
// channel gives 16 output times x 6 subbands over all 27 composed taps.
// Interior-only: t in [3, T-4] (boundary t handled by boundary_fix).
// =====================================================================
__global__ __launch_bounds__(32) void fused_frontend(
    const float* __restrict__ x,        // (B, C, T)
    const float* __restrict__ fir,      // (NB, 21)
    const float* __restrict__ w3,       // (NB,1,1,3)
    const float* __restrict__ w5,       // (NB,1,1,5)
    const float* __restrict__ w7,       // (NB,1,1,7)
    const float* __restrict__ bn1_g,
    const float* __restrict__ bn1_b,
    const float* __restrict__ bn1_m,
    const float* __restrict__ bn1_v,
    const float* __restrict__ sconv_w,  // (NB,1,C,1)
    float* __restrict__ hp)             // (B, NB, NBINS) pooled means
{
    const int lane   = threadIdx.x;
    const int tile   = blockIdx.x;
    const int b      = blockIdx.y;
    const int t0     = tile * 16;
    const int lmod   = lane & 15;       // M (time) row & N (subband) column
    const int hiHalf = lane >> 4;
    const bool valid = lmod < NB;       // only columns 0..5 carry real filters
    const int  nbc   = valid ? lmod : 0;

    __shared__ float xs[48];            // x[t0-13 .. t0+34] im2col staging
    __shared__ float hp_l[NB * 2];      // per-block pool partials (<=2 bins)

    // ---- effective 7-tap kernel: mean of the 3/5/7 depthwise convs ----
    float weff[7];
#pragma unroll
    for (int d = 0; d < 7; ++d) {
        float v = w7[nbc * 7 + d];
        if (d >= 1 && d <= 5) v += w5[nbc * 5 + (d - 1)];
        if (d >= 2 && d <= 4) v += w3[nbc * 3 + (d - 2)];
        weff[d] = valid ? v * (1.0f / 3.0f) : 0.0f;
    }

    // ---- composed 27-tap filter g = fir (*) weff -> B fragment ----
    // B (32x16 f16): lane holds column N=lmod, half h -> K = hiHalf*16 + h
    v16h bfrag;
#pragma unroll
    for (int h = 0; h < 16; ++h) {
        int k = hiHalf * 16 + h;
        float g = 0.0f;
        if (valid && k < GTAPS) {
#pragma unroll
            for (int d = 0; d < 7; ++d) {
                int i = k - d;
                if (i >= 0 && i < NTAPS) g += weff[d] * fir[nbc * NTAPS + i];
            }
        }
        bfrag[h] = (_Float16)g;
    }

    const float bnm    = bn1_m[nbc];
    const float bscale = bn1_g[nbc] * rsqrtf(bn1_v[nbc] + 1e-5f);
    const float bnb    = bn1_b[nbc];

    float acc[8];
#pragma unroll
    for (int v = 0; v < 8; ++v) acc[v] = 0.0f;

    const float* xrowBase = x + (size_t)b * CC * TT;
    const v8f czero = {0.f, 0.f, 0.f, 0.f, 0.f, 0.f, 0.f, 0.f};

#pragma clang loop unroll(disable)
    for (int c = 0; c < CC; ++c) {
        const float* xr = xrowBase + (size_t)c * TT;

        __syncthreads();   // previous iteration done reading xs
        {
            int tA = t0 - 13 + lane;
            xs[lane] = (tA >= 0 && tA < TT) ? xr[tA] : 0.0f;
            if (lane < 16) {
                int tB = tA + 32;
                xs[lane + 32] = (tB >= 0 && tB < TT) ? xr[tB] : 0.0f;
            }
        }
        __syncthreads();

        // A (16x32 f16): row M=lmod; half h -> K = hiHalf*16 + h
        v16h afrag;
#pragma unroll
        for (int h = 0; h < 16; ++h) {
            int k = hiHalf * 16 + h;
            afrag[h] = (_Float16)xs[lmod + k];   // = x[t0 + M + K - 13]
        }

        v8f dfrag = __builtin_amdgcn_wmma_f32_16x16x32_f16(
            false, afrag, false, bfrag, (short)0, czero, false, false);

        const float sw = valid ? sconv_w[nbc * CC + c] : 0.0f;
#pragma unroll
        for (int v = 0; v < 8; ++v) {
            float h1 = (dfrag[v] - bnm) * bscale + bnb;  // BN1
            h1 = fmaxf(h1, 0.0f);                        // ReLU
            acc[v] += h1 * sw;                           // sconv collapse
        }
    }

    // ---- mean-pool into NBINS bins; skip boundary t (exact kernel owns) ----
    __syncthreads();
    if (lane < NB * 2) hp_l[lane] = 0.0f;
    __syncthreads();

    const int bin0  = t0 / POOL;                 // uniform scalar
    const int bEdge = (bin0 + 1) * POOL;         // first t of next bin
    if (valid) {
#pragma unroll
        for (int v = 0; v < 8; ++v) {
            int t = t0 + v + hiHalf * 8;
            if (t >= 3 && t < TT - 3) {          // interior only
                int lb = (t >= bEdge) ? 1 : 0;
                atomicAdd(&hp_l[nbc * 2 + lb], acc[v]);
            }
        }
    }
    __syncthreads();
    if (lane < NB * 2) {
        int n2 = lane >> 1, lb = lane & 1, bin = bin0 + lb;
        if (bin < NBINS)
            atomicAdd(&hp[((size_t)b * NB + n2) * NBINS + bin],
                      hp_l[lane] * (1.0f / POOL));
    }
}

// =====================================================================
// Kernel 1b: exact h for the 6 boundary time samples per (b, nb) where
// the composed filter differs from the reference's xb zero-padding.
// One block per batch element; 36 work items (6 nb x 6 t). Rolled loops
// with computed bounds keep this kernel tiny.
// =====================================================================
__global__ __launch_bounds__(64) void boundary_fix(
    const float* __restrict__ x,
    const float* __restrict__ fir,
    const float* __restrict__ w3, const float* __restrict__ w5,
    const float* __restrict__ w7,
    const float* __restrict__ bn1_g, const float* __restrict__ bn1_b,
    const float* __restrict__ bn1_m, const float* __restrict__ bn1_v,
    const float* __restrict__ sconv_w,
    float* __restrict__ hp)
{
    const int b = blockIdx.x, tid = threadIdx.x;
    if (tid >= NB * 6) return;
    const int nb   = tid / 6;
    const int tIdx = tid % 6;
    const int t    = (tIdx < 3) ? tIdx : (TT - 6 + tIdx);  // {0,1,2,997,998,999}

    float weff[7];
#pragma unroll
    for (int d = 0; d < 7; ++d) {
        float v = w7[nb * 7 + d];
        if (d >= 1 && d <= 5) v += w5[nb * 5 + (d - 1)];
        if (d >= 2 && d <= 4) v += w3[nb * 3 + (d - 2)];
        weff[d] = v * (1.0f / 3.0f);
    }
    const float bnm    = bn1_m[nb];
    const float bscale = bn1_g[nb] * rsqrtf(bn1_v[nb] + 1e-5f);
    const float bnb    = bn1_b[nb];

    float acc = 0.0f;
#pragma clang loop unroll(disable)
    for (int c = 0; c < CC; ++c) {
        const float* xr = x + ((size_t)b * CC + c) * TT;
        float h1 = 0.0f;
#pragma clang loop unroll(disable)
        for (int d = 0; d < 7; ++d) {
            int tp = t + d - 3;
            if (tp >= 0 && tp < TT) {            // xb zero-padded outside [0,T)
                int iLo = (10 - tp > 0) ? (10 - tp) : 0;
                int iHi = (TT + 10 - tp < NTAPS) ? (TT + 10 - tp) : NTAPS;
                float xb = 0.0f;
#pragma clang loop unroll(disable)
                for (int i = iLo; i < iHi; ++i)
                    xb += fir[nb * NTAPS + i] * xr[tp + i - 10];
                h1 += weff[d] * xb;
            }
        }
        h1 = (h1 - bnm) * bscale + bnb;
        h1 = fmaxf(h1, 0.0f);
        acc += h1 * sconv_w[nb * CC + c];
    }
    const int bin = (t < POOL) ? 0 : (NBINS - 1);   // t is 0..2 or 997..999
    atomicAdd(&hp[((size_t)b * NB + nb) * NBINS + bin], acc * (1.0f / POOL));
}

// =====================================================================
// Kernel 2: gate (softmax + top-3 + renorm) + pooled experts + BN/ReLU +
// weighted combine + fc + log_softmax. One 64-thread block per batch.
// (125-mean-pool commuted through the linear expert einsum.)
// =====================================================================
__global__ __launch_bounds__(64) void moe_head(
    const float* __restrict__ hp,       // (B, NB, NBINS)
    const float* __restrict__ gate_w,   // (MNB, NB)
    const float* __restrict__ gate_b,   // (MNB)
    const float* __restrict__ exp_w,    // (E, MNB, NB)
    const float* __restrict__ exp_b,    // (E, MNB)
    const float* __restrict__ ebn_g, const float* __restrict__ ebn_b,
    const float* __restrict__ ebn_m, const float* __restrict__ ebn_v,
    const float* __restrict__ fc_w,     // (NCLS, MNB*NBINS)
    const float* __restrict__ fc_b,     // (NCLS)
    float* __restrict__ feats,          // (B, MNB*NBINS)
    float* __restrict__ logp)           // (B, NCLS)
{
    const int b = blockIdx.x, tid = threadIdx.x;
    __shared__ float sh_hp[NB * NBINS];
    __shared__ float sh_gw[MNB];
    __shared__ float sh_y[MNB * NBINS];
    __shared__ float sh_logit[NCLS];

    if (tid < NB * NBINS) sh_hp[tid] = hp[(size_t)b * NB * NBINS + tid];
    __syncthreads();

    if (tid < MNB) {
        float z = gate_b[tid];
        for (int nb = 0; nb < NB; ++nb) {
            float hm = 0.0f;
            for (int bin = 0; bin < NBINS; ++bin) hm += sh_hp[nb * NBINS + bin];
            z += (0.125f * hm) * gate_w[tid * NB + nb];
        }
        sh_gw[tid] = z;
    }
    __syncthreads();

    // top-3 + softmax restricted to top-3 (== softmax*mask/renorm)
    if (tid == 0) {
        int   i0 = 0, i1 = -1, i2 = -1;
        float v0 = -1e30f, v1 = -1e30f, v2 = -1e30f;
        for (int o = 0; o < MNB; ++o) {
            float z = sh_gw[o];
            if (z > v0)      { v2 = v1; i2 = i1; v1 = v0; i1 = i0; v0 = z; i0 = o; }
            else if (z > v1) { v2 = v1; i2 = i1; v1 = z; i1 = o; }
            else if (z > v2) { v2 = z; i2 = o; }
        }
        float e0 = 1.0f, e1 = expf(v1 - v0), e2 = expf(v2 - v0);
        float inv = 1.0f / (e0 + e1 + e2);
        for (int o = 0; o < MNB; ++o) {
            float g = 0.0f;
            if (o == i0) g = e0 * inv;
            else if (o == i1) g = e1 * inv;
            else if (o == i2) g = e2 * inv;
            sh_gw[o] = g;
        }
    }
    __syncthreads();

    for (int idx = tid; idx < MNB * NBINS; idx += 64) {
        int o = idx >> 3, bin = idx & 7;
        float val = 0.0f;
        for (int e = 0; e < EE; ++e) {
            int eo = e * MNB + o;
            float ye = exp_b[eo];
            for (int c = 0; c < NB; ++c)
                ye += sh_hp[c * NBINS + bin] * exp_w[eo * NB + c];
            float es = ebn_g[eo] * rsqrtf(ebn_v[eo] + 1e-5f);
            ye = (ye - ebn_m[eo]) * es + ebn_b[eo];
            ye = fmaxf(ye, 0.0f);
            val += sh_gw[e] * ye;        // gw[:, :E]
        }
        sh_y[idx] = val;
        feats[(size_t)b * MNB * NBINS + idx] = val;
    }
    __syncthreads();

    if (tid < NCLS) {
        float z = fc_b[tid];
#pragma clang loop unroll(disable)
        for (int j = 0; j < MNB * NBINS; ++j)
            z += sh_y[j] * fc_w[tid * (MNB * NBINS) + j];
        sh_logit[tid] = z;
    }
    __syncthreads();

    if (tid < NCLS) {
        float m = sh_logit[0];
        for (int k = 1; k < NCLS; ++k) m = fmaxf(m, sh_logit[k]);
        float s = 0.0f;
        for (int k = 0; k < NCLS; ++k) s += expf(sh_logit[k] - m);
        logp[(size_t)b * NCLS + tid] = sh_logit[tid] - m - logf(s);
    }
}

// =====================================================================
extern "C" void kernel_launch(void* const* d_in, const int* in_sizes, int n_in,
                              void* d_out, int out_size, void* d_ws, size_t ws_size,
                              hipStream_t stream) {
    const float* x       = (const float*)d_in[0];
    const float* fir     = (const float*)d_in[1];
    const float* w3      = (const float*)d_in[2];
    const float* w5      = (const float*)d_in[3];
    const float* w7      = (const float*)d_in[4];
    const float* bn1_g   = (const float*)d_in[5];
    const float* bn1_b   = (const float*)d_in[6];
    const float* bn1_m   = (const float*)d_in[7];
    const float* bn1_v   = (const float*)d_in[8];
    const float* sconv_w = (const float*)d_in[9];
    const float* gate_w  = (const float*)d_in[10];
    const float* gate_b  = (const float*)d_in[11];
    const float* exp_w   = (const float*)d_in[12];
    const float* exp_b   = (const float*)d_in[13];
    const float* ebn_g   = (const float*)d_in[14];
    const float* ebn_b   = (const float*)d_in[15];
    const float* ebn_m   = (const float*)d_in[16];
    const float* ebn_v   = (const float*)d_in[17];
    const float* fc_w    = (const float*)d_in[18];
    const float* fc_b    = (const float*)d_in[19];

    float* hp = (float*)d_ws;                              // (B, NB, NBINS)
    hipMemsetAsync(hp, 0, (size_t)BSZ * NB * NBINS * sizeof(float), stream);

    dim3 grid1(NTILES, BSZ);
    fused_frontend<<<grid1, 32, 0, stream>>>(
        x, fir, w3, w5, w7, bn1_g, bn1_b, bn1_m, bn1_v, sconv_w, hp);

    boundary_fix<<<BSZ, 64, 0, stream>>>(
        x, fir, w3, w5, w7, bn1_g, bn1_b, bn1_m, bn1_v, sconv_w, hp);

    float* feats = (float*)d_out;                          // (B, 384)
    float* logp  = feats + (size_t)BSZ * MNB * NBINS;      // (B, 9)
    moe_head<<<BSZ, 64, 0, stream>>>(
        hp, gate_w, gate_b, exp_w, exp_b, ebn_g, ebn_b, ebn_m, ebn_v,
        fc_w, fc_b, feats, logp);
}
